// WorldModel_74345883894060
// MI455X (gfx1250) — compile-verified, matrix operand order checked by reading, not run on hardware
//
#include <hip/hip_runtime.h>
#include <hip/hip_bf16.h>
#include <math.h>

// ---------------------------------------------------------------------------
// CDNA5 / gfx1250 world-model forward pass, v4.
// v_wmma_f32_16x16x32_f16 + Tensor Data Mover weight staging:
//  - all weights pre-converted to f16 in workspace (halves weight traffic)
//  - per K-chunk, wave0 issues tensor_load_to_lds for the NTx16 x 32 weight
//    tile (double-buffered, TDM zero-fills past tensor bounds -> no masking),
//    waits TENSORcnt, one barrier publishes; waves read B fragments with two
//    ds_load_b128 and issue NT back-to-back WMMAs per gathered A fragment.
// ---------------------------------------------------------------------------

typedef __attribute__((ext_vector_type(16))) _Float16 v16h;
typedef __attribute__((ext_vector_type(8)))  _Float16 v8h;
typedef __attribute__((ext_vector_type(8)))  float    v8f;
typedef __attribute__((ext_vector_type(4)))  unsigned int u32x4;
typedef __attribute__((ext_vector_type(8)))  int      i32x8;
typedef __attribute__((ext_vector_type(4)))  int      i32x4;

#define WAVES_PER_BLOCK 8

__device__ __forceinline__ float eluf(float x) {
  return x > 0.f ? x : (__expf(x) - 1.f);
}

__device__ __forceinline__ void tdm_load(u32x4 g0, i32x8 g1) {
  i32x4 z4 = {0, 0, 0, 0};
#if __clang_major__ >= 23
  i32x8 z8 = {0, 0, 0, 0, 0, 0, 0, 0};
  __builtin_amdgcn_tensor_load_to_lds(g0, g1, z4, z4, z8, 0);
#else
  __builtin_amdgcn_tensor_load_to_lds(g0, g1, z4, z4, 0);
#endif
}

#define GATHER_VIA_LOAD2                                                     \
  __device__ __forceinline__ void gather(int k0, int kh, float* av) const {  \
    _Pragma("unroll")                                                        \
    for (int i = 0; i < 8; ++i) {                                            \
      int kk = k0 + ((i < 4) ? (kh * 8 + i * 2)                              \
                             : (16 + kh * 8 + (i - 4) * 2));                 \
      load2(kk, av[2 * i], av[2 * i + 1]);                                   \
    }                                                                        \
  }

// ---- A-side loaders: init(m) once per tile, branchless gathers ------------

struct DenseA {  // K multiple of 32 in all uses (512); 16B-aligned rows
  const float* A; int M, K;
  const float* row; float mm;
  __device__ __forceinline__ void init(int m) {
    int mo = m < M ? m : M - 1;
    mm = (m < M) ? 1.f : 0.f;
    row = A + (size_t)mo * K;
  }
  __device__ __forceinline__ void gather(int k0, int kh, float* av) const {
    int kb = k0 + kh * 8;
    float4 x0 = *reinterpret_cast<const float4*>(row + kb);
    float4 x1 = *reinterpret_cast<const float4*>(row + kb + 4);
    float4 x2 = *reinterpret_cast<const float4*>(row + kb + 16);
    float4 x3 = *reinterpret_cast<const float4*>(row + kb + 20);
    av[0] = x0.x * mm;  av[1] = x0.y * mm;  av[2] = x0.z * mm;  av[3] = x0.w * mm;
    av[4] = x1.x * mm;  av[5] = x1.y * mm;  av[6] = x1.z * mm;  av[7] = x1.w * mm;
    av[8] = x2.x * mm;  av[9] = x2.y * mm;  av[10] = x2.z * mm; av[11] = x2.w * mm;
    av[12] = x3.x * mm; av[13] = x3.y * mm; av[14] = x3.z * mm; av[15] = x3.w * mm;
  }
};

struct ConcatA {  // [A1 | A2]; K1 multiple of 32 so pairs never straddle
  const float* A1; const float* A2; int M, K1, K2, stride2;
  const float* r1; const float* r2; float mm;
  __device__ __forceinline__ void init(int m) {
    int mo = m < M ? m : M - 1;
    mm = (m < M) ? 1.f : 0.f;
    r1 = A1 + (size_t)mo * K1;
    r2 = A2 + (size_t)mo * stride2;
  }
  __device__ __forceinline__ void load2(int k, float& f0, float& f1) const {
    float2 v; float mk0, mk1;
    if (k < K1) {                                  // uniform per 32-K chunk
      v = *reinterpret_cast<const float2*>(r1 + k);
      mk0 = mk1 = mm;
    } else {
      int k2 = k - K1;
      int kc = (k2 + 1 < K2) ? k2 : (K2 - 2);     // K2 even (6, 512, 1024)
      v = *reinterpret_cast<const float2*>(r2 + kc);
      mk0 = (k2 < K2) ? mm : 0.f;
      mk1 = (k2 + 1 < K2) ? mm : 0.f;
    }
    f0 = v.x * mk0; f1 = v.y * mk1;
  }
  GATHER_VIA_LOAD2
};

// encoder L0: obs [B=32,T=64,3,64,64] NCHW, image order (t,b); k4 s2 VALID
struct Im2colObsA {
  const float* X; int img0; int M;
  const float* base; float mm;
  __device__ __forceinline__ void init(int m) {
    int mo = m < M ? m : M - 1;
    mm = (m < M) ? 1.f : 0.f;
    int img = img0 + mo / 961;
    int rem = mo % 961;
    int oy = rem / 31, ox = rem % 31;
    int t = img >> 5, b = img & 31;
    base = X + (size_t)(b * 64 + t) * 12288 + (size_t)(oy * 2) * 64 + (ox * 2);
  }
  __device__ __forceinline__ void load2(int k, float& f0, float& f1) const {
    int kc = (k < 48) ? k : 46;                    // 48 even: pair all-in/out
    float mk = (k < 48) ? mm : 0.f;
    int ci = kc >> 4, kr = kc & 15;                // KS=4
    const float* p = base + ci * 4096 + (kr >> 2) * 64 + (kr & 3);
    f0 = p[0] * mk; f1 = p[1] * mk;
  }
  GATHER_VIA_LOAD2
};

// encoder L1-3: X [img][H][W][C] NHWC, k4 s2 VALID; all taps in-bounds
struct Im2colNhwcA {
  const float* X; int M, K, C, H, W, OW, OHW;
  const float* base; float mm;
  __device__ __forceinline__ void init(int m) {
    int mo = m < M ? m : M - 1;
    mm = (m < M) ? 1.f : 0.f;
    int img = mo / OHW, rem = mo % OHW;
    int oy = rem / OW, ox = rem % OW;
    base = X + (((size_t)img * H + oy * 2) * W + ox * 2) * C;
  }
  __device__ __forceinline__ void load2(int k, float& f0, float& f1) const {
    int kc = (k + 1 < K) ? k : (K - 2);
    float mk0 = (k < K) ? mm : 0.f, mk1 = (k + 1 < K) ? mm : 0.f;
    int ci = kc >> 4, kr = kc & 15;                // KS=4
    int ky = kr >> 2, kx = kr & 3;                 // kx even -> kx+1 same row
    const float* p = base + ((size_t)ky * W + kx) * C + ci;
    f0 = p[0] * mk0; f1 = p[C] * mk1;
  }
  GATHER_VIA_LOAD2
};

// conv_transpose stride 2 VALID, X [img][IH][IW][C] NHWC; KS compile-time
template <int KS>
struct DeconvA {
  const float* X; int M, K, C, IH, IW, OW, OHW;
  const float* base; int oy, ox; float mm;
  __device__ __forceinline__ void init(int m) {
    int mo = m < M ? m : M - 1;
    mm = (m < M) ? 1.f : 0.f;
    int img = mo / OHW, rem = mo % OHW;
    oy = rem / OW; ox = rem % OW;
    base = X + (size_t)img * IH * IW * C;
  }
  __device__ __forceinline__ float ld(int k) const {
    int kc = (k < K) ? k : (K - 1);
    float mk = (k < K) ? mm : 0.f;
    int ci = kc / (KS * KS), kr = kc % (KS * KS);
    int ky = kr / KS, kx = kr % KS;
    int ty = oy - ky, tx = ox - kx;
    int iy = ty >> 1, ix = tx >> 1;
    bool valid = (ty >= 0) && (tx >= 0) && !(ty & 1) && !(tx & 1) &&
                 (iy < IH) && (ix < IW);
    int iyc = valid ? iy : 0, ixc = valid ? ix : 0;
    float v = base[((size_t)iyc * IW + ixc) * C + ci];
    return valid ? v * mk : 0.f;
  }
  __device__ __forceinline__ void load2(int k, float& f0, float& f1) const {
    f0 = ld(k); f1 = ld(k + 1);
  }
  GATHER_VIA_LOAD2
};

// ---- WMMA GEMM: C[M,N] = act(A[M,K] @ W[N,K]^T + bias), W in f16 ----------
// Block = 8 waves x 1 M-tile, NT N-tiles (register blocking). Weight tile
// [NT*16 rows x 32 K] DMA'd to LDS by the TDM (wave0 issues, TENSORcnt wait,
// one barrier per chunk, double-buffered; TDM zero-fills OOB rows/cols).
template <class AL, int NT>
__global__ void gemm_bias_act(AL al, const _Float16* __restrict__ Wt,
                              const float* __restrict__ bias,
                              float* __restrict__ Co,
                              int M, int N, int K, int act) {
  __shared__ __align__(64) _Float16 sW[2][NT * 512];  // [row][k] row-major
  int tid = threadIdx.x;
  int wave = tid >> 5, lane = tid & 31;
  int kh = lane >> 4, l16 = lane & 15;
  int tilesM = (M + 15) >> 4;
  int tm = blockIdx.x * WAVES_PER_BLOCK + wave;
  int n0 = blockIdx.y * (NT * 16);
  bool active = (tm < tilesM);
  int m = (tm << 4) + l16;
  if (active) al.init(m);

  auto issue_tdm = [&](int k0w, int buf) {
    uint32_t lds_off = (uint32_t)(uintptr_t)(&sW[buf][0]);
    uint64_t ga = (uint64_t)(uintptr_t)(Wt + (size_t)n0 * K + k0w);
    uint32_t td0 = (uint32_t)(K - k0w);   // remaining K (halfs) -> zero-fill
    uint32_t td1 = (uint32_t)(N - n0);    // remaining rows      -> zero-fill
    u32x4 g0 = {1u, lds_off, (uint32_t)ga,
                (uint32_t)((ga >> 32) & 0x01FFFFFFu) | 0x80000000u};
    i32x8 g1 = {(int)(1u << 16),                                  // 2B elems
                (int)((td0 & 0xFFFFu) << 16),
                (int)(((td0 >> 16) & 0xFFFFu) | ((td1 & 0xFFFFu) << 16)),
                (int)(((td1 >> 16) & 0xFFFFu) | (32u << 16)),     // tile_dim0
                (int)(NT * 16),                                   // tile_dim1
                (int)(uint32_t)K,                                 // row stride
                0, 0};
    tdm_load(g0, g1);
  };

  v8f acc[NT];
#pragma unroll
  for (int nt = 0; nt < NT; ++nt) acc[nt] = v8f{};

  if (wave == 0) issue_tdm(0, 0);
  int bufi = 0;
  for (int k0 = 0; k0 < K; k0 += 32) {
    if (wave == 0) __builtin_amdgcn_s_wait_tensorcnt(0);
    __syncthreads();                       // tile bufi ready for all waves
    if (wave == 0 && k0 + 32 < K) issue_tdm(k0 + 32, bufi ^ 1);
    if (active) {
      float av[16];
      al.gather(k0, kh, av);
      v16h a;
#pragma unroll
      for (int j = 0; j < 16; ++j) a[j] = (_Float16)av[j];
#pragma unroll
      for (int nt = 0; nt < NT; ++nt) {
        int rb = (nt * 16 + l16) * 32;
        v8h blo = *reinterpret_cast<const v8h*>(&sW[bufi][rb + kh * 8]);
        v8h bhi = *reinterpret_cast<const v8h*>(&sW[bufi][rb + 16 + kh * 8]);
        v16h b = __builtin_shufflevector(blo, bhi, 0, 1, 2, 3, 4, 5, 6, 7,
                                         8, 9, 10, 11, 12, 13, 14, 15);
        acc[nt] = __builtin_amdgcn_wmma_f32_16x16x32_f16(
            false, a, false, b, (short)0, acc[nt], false, false);
      }
    }
    bufi ^= 1;
  }

  if (active) {
#pragma unroll
    for (int nt = 0; nt < NT; ++nt) {
      int n = n0 + nt * 16 + l16;
#pragma unroll
      for (int r = 0; r < 8; ++r) {
        int mr = (tm << 4) + kh * 8 + r;
        if (mr < M && n < N) {
          float v = acc[nt][r] + bias[n];
          if (act) v = eluf(v);
          Co[(size_t)mr * N + n] = v;
        }
      }
    }
  }
}

// ---- elementwise kernels --------------------------------------------------

__global__ void cvt_f16(const float* __restrict__ src,
                        _Float16* __restrict__ dst, int n) {
  int i = blockIdx.x * blockDim.x + threadIdx.x;
  if (i < n) dst[i] = (_Float16)src[i];
}

__global__ void permute_emb(const float* __restrict__ src,
                            float* __restrict__ dst, int nimg) {
  int idx = blockIdx.x * blockDim.x + threadIdx.x;
  if (idx >= nimg * 1024) return;
  int img = idx >> 10, r = idx & 1023;
  int c = r >> 2, s4 = r & 3;
  dst[(size_t)img * 1024 + r] = src[(size_t)img * 1024 + s4 * 256 + c];
}

__global__ void gru_combine(const float* __restrict__ gi,
                            const float* __restrict__ gh,
                            const float* __restrict__ hprev,
                            float* __restrict__ hcur,
                            float* __restrict__ hs_out) {
  int idx = blockIdx.x * blockDim.x + threadIdx.x;
  if (idx >= 32 * 512) return;
  int b = idx >> 9, j = idx & 511;
  const float* gib = gi + (size_t)b * 1536;
  const float* ghb = gh + (size_t)b * 1536;
  float ir = gib[j], iz = gib[512 + j], in = gib[1024 + j];
  float hr = ghb[j], hz = ghb[512 + j], hn = ghb[1024 + j];
  float r  = 1.f / (1.f + __expf(-(ir + hr)));
  float u  = 1.f / (1.f + __expf(-(iz + hz)));
  float nn = tanhf(in + r * hn);
  float hv = hprev[idx];
  float hnew = (1.f - u) * nn + u * hv;
  hcur[idx] = hnew;
  hs_out[idx] = hnew;
}

__global__ void sample_kl(const float* __restrict__ pos,
                          const float* __restrict__ pri,
                          const float* __restrict__ gum,
                          float* __restrict__ zs_ws,
                          float* __restrict__ zs_out,
                          float* __restrict__ klacc) {
  int row = blockIdx.x * blockDim.x + threadIdx.x;
  if (row >= 1024) return;
  const float* pr = pos + (size_t)row * 32;
  const float* qr = pri + (size_t)row * 32;
  const float* gr = gum + (size_t)row * 32;
  float mp = -1e30f, mq = -1e30f, mg = -1e30f; int amax = 0;
  for (int k = 0; k < 32; ++k) {
    float p = pr[k];
    mp = fmaxf(mp, p);
    mq = fmaxf(mq, qr[k]);
    float pg = p + gr[k];
    if (pg > mg) { mg = pg; amax = k; }
  }
  float sp = 0.f, sq = 0.f;
  for (int k = 0; k < 32; ++k) { sp += __expf(pr[k] - mp); sq += __expf(qr[k] - mq); }
  float lzp = mp + __logf(sp), lzq = mq + __logf(sq);
  float kl = 0.f;
  for (int k = 0; k < 32; ++k) {
    float lp = pr[k] - lzp, lq = qr[k] - lzq;
    kl += __expf(lp) * (lp - lq);
    float zv = (k == amax) ? 1.f : 0.f;
    zs_ws[(size_t)row * 32 + k] = zv;
    zs_out[(size_t)row * 32 + k] = zv;
  }
  atomicAdd(klacc, kl * (1.f / 1024.f));
}

__global__ void loss_accum(const float* __restrict__ mean,
                           const float* __restrict__ obs,
                           int start, int S, float* __restrict__ acc) {
  __shared__ float sh[256];
  const float LOG2PI = 1.8378770664093453f;
  long total = (long)S * 12288;
  float local = 0.f;
  for (long e = blockIdx.x * (long)blockDim.x + threadIdx.x; e < total;
       e += (long)gridDim.x * blockDim.x) {
    int sl = (int)(e / 12288);
    int rem = (int)(e % 12288);
    int c = rem / 4096, pix = rem % 4096;
    float mv = mean[((size_t)sl * 4096 + pix) * 3 + c];
    int s = start + sl;
    int ts = s >> 5, b = s & 31;
    float tv = obs[(((size_t)b * 64 + (ts + 1)) * 3 + c) * 4096 + pix];
    float d = tv - mv;
    local += 0.5f * (d * d + LOG2PI);
  }
  sh[threadIdx.x] = local;
  __syncthreads();
  for (int o = 128; o > 0; o >>= 1) {
    if (threadIdx.x < o) sh[threadIdx.x] += sh[threadIdx.x + o];
    __syncthreads();
  }
  if (threadIdx.x == 0) atomicAdd(acc, sh[0]);
}

__global__ void finalize_loss(const float* __restrict__ acc,
                              const float* __restrict__ kls,
                              float* __restrict__ out0) {
  if (threadIdx.x == 0 && blockIdx.x == 0) {
    float ks = 0.f;
    for (int t = 0; t < 63; ++t) ks += kls[t];
    out0[0] = acc[0] / 2016.f + 0.1f * (ks / 63.f);
  }
}

// ---- workspace layout (float offsets; every region 32B-aligned) -----------
enum : size_t {
  OF_EMBS = 0,
  OF_ENCA = OF_EMBS + (size_t)2048 * 1024,
  OF_ENCB = OF_ENCA + (size_t)256 * 961 * 32,
  OF_ZS   = OF_ENCB + (size_t)256 * 196 * 64,
  OF_HS   = OF_ZS   + (size_t)63 * 32 * 1024,
  OF_Z0   = OF_HS   + (size_t)63 * 32 * 512,
  OF_H0   = OF_Z0   + 32 * 1024,
  OF_X    = OF_H0   + 32 * 512,
  OF_GI   = OF_X    + 32 * 512,
  OF_GH   = OF_GI   + 32 * 1536,
  OF_H1   = OF_GH   + 32 * 1536,
  OF_PRI  = OF_H1   + 32 * 512,
  OF_H2   = OF_PRI  + 32 * 1024,
  OF_POS  = OF_H2   + 32 * 512,
  OF_KLS  = OF_POS  + 32 * 1024,
  OF_LACC = OF_KLS  + 64,
  OF_DLIN = OF_LACC + 64,
  OF_D0   = OF_DLIN + (size_t)224 * 1024,
  OF_D1   = OF_D0   + (size_t)224 * 25 * 128,
  OF_D2   = OF_D1   + (size_t)224 * 169 * 64,
  OF_D3   = OF_D2   + (size_t)224 * 900 * 32,
  OF_WH   = OF_D3   + (size_t)224 * 4096 * 3     // f16 weight arena (below)
};

// f16 weight arena offsets (in halfs)
enum : size_t {
  WH_ENC0 = 0,
  WH_ENC1 = WH_ENC0 + (size_t)32 * 48,
  WH_ENC2 = WH_ENC1 + (size_t)64 * 512,
  WH_ENC3 = WH_ENC2 + (size_t)128 * 1024,
  WH_ZA   = WH_ENC3 + (size_t)256 * 2048,
  WH_GWI  = WH_ZA   + (size_t)512 * 1030,
  WH_GWH  = WH_GWI  + (size_t)1536 * 512,
  WH_PRI0 = WH_GWH  + (size_t)1536 * 512,
  WH_PRIO = WH_PRI0 + (size_t)512 * 512,
  WH_POS0 = WH_PRIO + (size_t)1024 * 512,
  WH_POSO = WH_POS0 + (size_t)512 * 1536,
  WH_DLIN = WH_POSO + (size_t)1024 * 512,
  WH_D0   = WH_DLIN + (size_t)1024 * 1536,
  WH_D1   = WH_D0   + (size_t)128 * 25600,
  WH_D2   = WH_D1   + (size_t)64 * 3200,
  WH_D3   = WH_D2   + (size_t)32 * 2304,
  WH_END  = WH_D3   + (size_t)3 * 1152
};

template <int NT, class AL>
static inline void run_gemm(const AL& al, const _Float16* W, const float* b,
                            float* C, int M, int N, int K, int act,
                            hipStream_t s) {
  int tilesM = (M + 15) / 16;
  int ngrp = (N + 16 * NT - 1) / (16 * NT);
  dim3 grid((tilesM + WAVES_PER_BLOCK - 1) / WAVES_PER_BLOCK, ngrp);
  gemm_bias_act<AL, NT><<<grid, 32 * WAVES_PER_BLOCK, 0, s>>>(al, W, b, C,
                                                              M, N, K, act);
}

static inline void cvt_w(const float* s, _Float16* d, int n, hipStream_t st) {
  cvt_f16<<<(n + 255) / 256, 256, 0, st>>>(s, d, n);
}

extern "C" void kernel_launch(void* const* d_in, const int* in_sizes, int n_in,
                              void* d_out, int out_size, void* d_ws, size_t ws_size,
                              hipStream_t stream) {
  (void)in_sizes; (void)n_in; (void)out_size; (void)ws_size;
  const float* obs     = (const float*)d_in[0];   // [32,64,3,64,64]
  const float* actions = (const float*)d_in[1];   // [32,64,6]
  const float* gumbel  = (const float*)d_in[2];   // [63,32,32,32]
  const float* enc_w[4]; const float* enc_b[4];
  for (int i = 0; i < 4; ++i) {
    enc_w[i] = (const float*)d_in[3 + 2 * i];
    enc_b[i] = (const float*)d_in[4 + 2 * i];
  }
  const float* za_w0     = (const float*)d_in[11];
  const float* za_b0     = (const float*)d_in[12];
  const float* gru_wi    = (const float*)d_in[13];
  const float* gru_bi    = (const float*)d_in[14];
  const float* gru_wh    = (const float*)d_in[15];
  const float* gru_bh    = (const float*)d_in[16];
  const float* pri_w0    = (const float*)d_in[17];
  const float* pri_b0    = (const float*)d_in[18];
  const float* pri_wo    = (const float*)d_in[19];
  const float* pri_bo    = (const float*)d_in[20];
  const float* pos_w0    = (const float*)d_in[21];
  const float* pos_b0    = (const float*)d_in[22];
  const float* pos_wo    = (const float*)d_in[23];
  const float* pos_bo    = (const float*)d_in[24];
  const float* dec_lin_w = (const float*)d_in[25];
  const float* dec_lin_b = (const float*)d_in[26];
  const float* dec_w[4]; const float* dec_b[4];
  for (int i = 0; i < 4; ++i) {
    dec_w[i] = (const float*)d_in[27 + 2 * i];
    dec_b[i] = (const float*)d_in[28 + 2 * i];
  }

  float* ws   = (float*)d_ws;
  float* embs = ws + OF_EMBS;
  float* encA = ws + OF_ENCA;
  float* encB = ws + OF_ENCB;
  float* ZS   = ws + OF_ZS;
  float* HS   = ws + OF_HS;
  float* Z0   = ws + OF_Z0;
  float* H0   = ws + OF_H0;
  float* X    = ws + OF_X;
  float* GI   = ws + OF_GI;
  float* GH   = ws + OF_GH;
  float* H1   = ws + OF_H1;
  float* PRI  = ws + OF_PRI;
  float* H2   = ws + OF_H2;
  float* POS  = ws + OF_POS;
  float* KLS  = ws + OF_KLS;
  float* LACC = ws + OF_LACC;
  float* DLIN = ws + OF_DLIN;
  float* D0   = ws + OF_D0;
  float* D1   = ws + OF_D1;
  float* D2   = ws + OF_D2;
  float* D3   = ws + OF_D3;
  _Float16* WH = (_Float16*)(ws + OF_WH);

  float* out    = (float*)d_out;
  float* zs_out = out + 1;                           // [63,32,1024]
  float* hs_out = out + 1 + (size_t)63 * 32 * 1024;  // [63,32,512]

  hipMemsetAsync(Z0, 0, (32 * 1024 + 32 * 512) * sizeof(float), stream);
  hipMemsetAsync(KLS, 0, 128 * sizeof(float), stream);

  // ---- one-time (per call) weight f32 -> f16 conversion -------------------
  cvt_w(enc_w[0], WH + WH_ENC0, 32 * 48, stream);
  cvt_w(enc_w[1], WH + WH_ENC1, 64 * 512, stream);
  cvt_w(enc_w[2], WH + WH_ENC2, 128 * 1024, stream);
  cvt_w(enc_w[3], WH + WH_ENC3, 256 * 2048, stream);
  cvt_w(za_w0,    WH + WH_ZA,   512 * 1030, stream);
  cvt_w(gru_wi,   WH + WH_GWI,  1536 * 512, stream);
  cvt_w(gru_wh,   WH + WH_GWH,  1536 * 512, stream);
  cvt_w(pri_w0,   WH + WH_PRI0, 512 * 512, stream);
  cvt_w(pri_wo,   WH + WH_PRIO, 1024 * 512, stream);
  cvt_w(pos_w0,   WH + WH_POS0, 512 * 1536, stream);
  cvt_w(pos_wo,   WH + WH_POSO, 1024 * 512, stream);
  cvt_w(dec_lin_w, WH + WH_DLIN, 1024 * 1536, stream);
  cvt_w(dec_w[0], WH + WH_D0, 128 * 25600, stream);
  cvt_w(dec_w[1], WH + WH_D1, 64 * 3200, stream);
  cvt_w(dec_w[2], WH + WH_D2, 32 * 2304, stream);
  cvt_w(dec_w[3], WH + WH_D3, 3 * 1152, stream);

  // ---- encoder: 2048 images (t-major), 8 chunks of 256 --------------------
  for (int c = 0; c < 8; ++c) {
    int g0 = c * 256;
    Im2colObsA a0{obs, g0, 256 * 961};
    run_gemm<2>(a0, WH + WH_ENC0, enc_b[0], encA, 256 * 961, 32, 48, 1, stream);
    Im2colNhwcA a1{encA, 256 * 196, 512, 32, 31, 31, 14, 196};
    run_gemm<4>(a1, WH + WH_ENC1, enc_b[1], encB, 256 * 196, 64, 512, 1, stream);
    Im2colNhwcA a2{encB, 256 * 36, 1024, 64, 14, 14, 6, 36};
    run_gemm<4>(a2, WH + WH_ENC2, enc_b[2], encA, 256 * 36, 128, 1024, 1, stream);
    Im2colNhwcA a3{encA, 256 * 4, 2048, 128, 6, 6, 2, 4};
    run_gemm<4>(a3, WH + WH_ENC3, enc_b[3], encB, 256 * 4, 256, 2048, 1, stream);
    permute_emb<<<(256 * 1024 + 255) / 256, 256, 0, stream>>>(
        encB, embs + (size_t)g0 * 1024, 256);
  }

  // ---- recurrent rollout: 63 steps ----------------------------------------
  for (int t = 0; t < 63; ++t) {
    const float* zprev = (t == 0) ? Z0 : (ZS + (size_t)(t - 1) * 32768);
    const float* hprev = (t == 0) ? H0 : (HS + (size_t)(t - 1) * 16384);
    float* hcur = HS + (size_t)t * 16384;
    ConcatA ax{zprev, actions + (size_t)t * 6, 32, 1024, 6, 64 * 6};
    run_gemm<4>(ax, WH + WH_ZA, za_b0, X, 32, 512, 1030, 1, stream);
    DenseA dgi{X, 32, 512};
    run_gemm<4>(dgi, WH + WH_GWI, gru_bi, GI, 32, 1536, 512, 0, stream);
    DenseA dgh{hprev, 32, 512};
    run_gemm<4>(dgh, WH + WH_GWH, gru_bh, GH, 32, 1536, 512, 0, stream);
    gru_combine<<<64, 256, 0, stream>>>(GI, GH, hprev, hcur,
                                        hs_out + (size_t)t * 16384);
    DenseA dh{hcur, 32, 512};
    run_gemm<4>(dh, WH + WH_PRI0, pri_b0, H1, 32, 512, 512, 1, stream);
    DenseA dh1{H1, 32, 512};
    run_gemm<4>(dh1, WH + WH_PRIO, pri_bo, PRI, 32, 1024, 512, 0, stream);
    ConcatA ap{hcur, embs + (size_t)(t + 1) * 32768, 32, 512, 1024, 1024};
    run_gemm<4>(ap, WH + WH_POS0, pos_b0, H2, 32, 512, 1536, 1, stream);
    DenseA dh2{H2, 32, 512};
    run_gemm<4>(dh2, WH + WH_POSO, pos_bo, POS, 32, 1024, 512, 0, stream);
    sample_kl<<<4, 256, 0, stream>>>(POS, PRI, gumbel + (size_t)t * 32768,
                                     ZS + (size_t)t * 32768,
                                     zs_out + (size_t)t * 32768, KLS + t);
  }

  // ---- decoder + loss: 2016 samples, 9 chunks of 224 (M always /16) -------
  for (int c = 0; c < 9; ++c) {
    int s0 = c * 224;
    ConcatA adl{ZS + (size_t)s0 * 1024, HS + (size_t)s0 * 512,
                224, 1024, 512, 512};
    run_gemm<4>(adl, WH + WH_DLIN, dec_lin_b, DLIN, 224, 1024, 1536, 0, stream);
    DeconvA<5> a0{DLIN, 224 * 25, 25600, 1024, 1, 1, 5, 25};
    run_gemm<4>(a0, WH + WH_D0, dec_b[0], D0, 224 * 25, 128, 25600, 1, stream);
    DeconvA<5> a1{D0, 224 * 169, 3200, 128, 5, 5, 13, 169};
    run_gemm<4>(a1, WH + WH_D1, dec_b[1], D1, 224 * 169, 64, 3200, 1, stream);
    DeconvA<6> a2{D1, 224 * 900, 2304, 64, 13, 13, 30, 900};
    run_gemm<2>(a2, WH + WH_D2, dec_b[2], D2, 224 * 900, 32, 2304, 1, stream);
    DeconvA<6> a3{D2, 224 * 4096, 1152, 32, 30, 30, 64, 4096};
    run_gemm<1>(a3, WH + WH_D3, dec_b[3], D3, 224 * 4096, 3, 1152, 0, stream);
    loss_accum<<<512, 256, 0, stream>>>(D3, obs, s0, 224, LACC);
  }

  finalize_loss<<<1, 32, 0, stream>>>(LACC, KLS, out);
}
